// EdgeAwareAttention_25039659336345
// MI455X (gfx1250) — compile-verified, matrix operand order checked by reading, not run on hardware
//
#include <hip/hip_runtime.h>
#include <hip/hip_bf16.h>
#include <math.h>

// ---------------- problem constants ----------------
#define BN 2
#define NN 2048
#define HH 256
#define NHEAD 8
#define HD 32
#define EDGES 65536
#define ROWS (BN * NN)          // 4096
#define CAP 256                 // per-row neighbor capacity (avg deg = 32)
#define LN_EPS 1e-5f

typedef __attribute__((ext_vector_type(8)))  __bf16 v8bf;
typedef __attribute__((ext_vector_type(16))) __bf16 v16bf;
typedef __attribute__((ext_vector_type(8)))  float  v8f;

// ---------------- fp32 -> bf16 convert (grid-stride) ----------------
__global__ __launch_bounds__(256) void cvt_bf16_kernel(const float* __restrict__ src,
                                                       __bf16* __restrict__ dst, int n) {
  for (int i = blockIdx.x * blockDim.x + threadIdx.x; i < n; i += gridDim.x * blockDim.x)
    dst[i] = (__bf16)src[i];
}

// ---------------- zero u32 (mask bits + degree) ----------------
__global__ __launch_bounds__(256) void zero_kernel(unsigned int* __restrict__ p, int n) {
  for (int i = blockIdx.x * blockDim.x + threadIdx.x; i < n; i += gridDim.x * blockDim.x)
    p[i] = 0u;
}

// ---------------- edge list -> dedup bitmask + capped CSR ----------------
__global__ __launch_bounds__(256) void edge_kernel(const int* __restrict__ ei,
                                                   unsigned int* __restrict__ maskbits,
                                                   int* __restrict__ deg,
                                                   int* __restrict__ adj) {
  int e = blockIdx.x * blockDim.x + threadIdx.x;
  if (e >= EDGES) return;
  int src = ei[e];
  int dst = ei[EDGES + e];
  unsigned int bit = 1u << (dst & 31);
  unsigned int old = atomicOr(&maskbits[src * (NN / 32) + (dst >> 5)], bit);
  if (!(old & bit)) {
    int p = atomicAdd(&deg[src], 1);
    if (p < CAP) adj[src * CAP + p] = dst;   // overflow beyond CAP statistically impossible; dropped
  }
}

// ---------------- bf16 WMMA GEMM: C[m][n] = sum_k A[m][k]*Wt[n][k] (+resid) ----------------
// One wave = one 16x16 tile. Layout per CDNA5 ISA 7.12.2 (16-bit A 16x32 / B 32x16):
//   lane<16 : row/col = lane, K in {ksel*8 .. +7} U {ksel*8+16 .. +23} with ksel=0
//   lane>=16: same row/col, ksel=1  -> per-lane two contiguous 16B loads.
__global__ __launch_bounds__(256) void gemm_bf16_wmma_kernel(const __bf16* __restrict__ A,
                                                             const __bf16* __restrict__ Wt,
                                                             const float* __restrict__ resid,
                                                             float* __restrict__ C,
                                                             int Nout, int K) {
  int wave = (int)((blockIdx.x * blockDim.x + threadIdx.x) >> 5);
  int lane = threadIdx.x & 31;
  int ntn  = Nout >> 4;
  int tm   = (wave / ntn) << 4;
  int tn   = (wave % ntn) << 4;
  int ksel = lane >> 4;            // 0 or 1
  int r    = lane & 15;

  const __bf16* pa = A  + (size_t)(tm + r) * K + ksel * 8;
  const __bf16* pb = Wt + (size_t)(tn + r) * K + ksel * 8;

  v8f acc = {};
  for (int k0 = 0; k0 < K; k0 += 32) {
    v8bf alo = *(const v8bf*)(pa + k0);
    v8bf ahi = *(const v8bf*)(pa + k0 + 16);
    v8bf blo = *(const v8bf*)(pb + k0);
    v8bf bhi = *(const v8bf*)(pb + k0 + 16);
    v16bf af = __builtin_shufflevector(alo, ahi, 0,1,2,3,4,5,6,7,8,9,10,11,12,13,14,15);
    v16bf bf = __builtin_shufflevector(blo, bhi, 0,1,2,3,4,5,6,7,8,9,10,11,12,13,14,15);
    acc = __builtin_amdgcn_wmma_f32_16x16x32_bf16(false, af, false, bf, (short)0, acc,
                                                  false, false);
  }

  // C layout (32-bit 16x16): lane<16 -> M=i,N=lane ; lane>=16 -> M=8+i,N=lane-16
  int n  = tn + r;
  int mb = tm + ksel * 8;
#pragma unroll
  for (int i = 0; i < 8; ++i) {
    float v = acc[i];
    size_t o = (size_t)(mb + i) * Nout + n;
    if (resid) v += resid[o];
    C[o] = v;
  }
}

// ---------------- sparse masked attention: one wave per (b, head, row) ----------------
// QKV layout: qkv[(b*NN+node)*768 + {0|256|512} + head*32 + d]
__global__ __launch_bounds__(256) void attn_kernel(const float* __restrict__ qkv,
                                                   const int* __restrict__ deg,
                                                   const int* __restrict__ adj,
                                                   __bf16* __restrict__ attnbf) {
  __shared__ float lds_q[8][32];
  __shared__ float lds_s[8][CAP];

  int w    = threadIdx.x >> 5;
  int lane = threadIdx.x & 31;
  int wg   = blockIdx.x * 8 + w;                 // global wave id
  int b    = wg / (NHEAD * NN);
  int rem  = wg % (NHEAD * NN);
  int h    = rem / NN;
  int row  = rem % NN;

  size_t obase = (size_t)(b * NN + row) * HH + h * HD + lane;
  int d = deg[row];
  if (d > CAP) d = CAP;
  if (d == 0) { attnbf[obase] = (__bf16)0.0f; return; }

  lds_q[w][lane] = qkv[(size_t)(b * NN + row) * 768 + h * HD + lane];
  const float scale = 0.1767766952966369f;       // 1/sqrt(32)

  // pass 1: scores (lane-per-neighbor)
  for (int base = 0; base < d; base += 32) {
    int idx = base + lane;
    if (idx < d) {
      int j = adj[row * CAP + idx];
      const float* kp = qkv + (size_t)(b * NN + j) * 768 + 256 + h * HD;
      float s = 0.0f;
#pragma unroll
      for (int t = 0; t < HD; ++t) s += lds_q[w][t] * kp[t];
      lds_s[w][idx] = s * scale;
    }
  }
  // row max
  float m = -3.0e38f;
  for (int idx = lane; idx < d; idx += 32) m = fmaxf(m, lds_s[w][idx]);
  for (int o = 16; o > 0; o >>= 1) m = fmaxf(m, __shfl_xor(m, o, 32));
  // exp + sum
  float ssum = 0.0f;
  for (int idx = lane; idx < d; idx += 32) {
    float e = __expf(lds_s[w][idx] - m);
    lds_s[w][idx] = e;
    ssum += e;
  }
  for (int o = 16; o > 0; o >>= 1) ssum += __shfl_xor(ssum, o, 32);
  // weighted V accumulation (lane = dim)
  float acc = 0.0f;
  for (int t = 0; t < d; ++t) {
    float wgt = lds_s[w][t];
    int j = adj[row * CAP + t];
    acc += wgt * qkv[(size_t)(b * NN + j) * 768 + 512 + h * HD + lane];
  }
  attnbf[obase] = (__bf16)(acc / ssum);
}

// ---------------- LayerNorm: one wave per row of 256 ----------------
__global__ __launch_bounds__(256) void ln_kernel(const float* __restrict__ y,
                                                 const float* __restrict__ gamma,
                                                 const float* __restrict__ beta,
                                                 float* __restrict__ out) {
  int w    = threadIdx.x >> 5;
  int lane = threadIdx.x & 31;
  int row  = blockIdx.x * 8 + w;

  float v[8];
  float sum = 0.0f;
#pragma unroll
  for (int i = 0; i < 8; ++i) {
    v[i] = y[(size_t)row * HH + i * 32 + lane];
    sum += v[i];
  }
  for (int o = 16; o > 0; o >>= 1) sum += __shfl_xor(sum, o, 32);
  float mu = sum * (1.0f / HH);
  float vs = 0.0f;
#pragma unroll
  for (int i = 0; i < 8; ++i) { float t = v[i] - mu; vs += t * t; }
  for (int o = 16; o > 0; o >>= 1) vs += __shfl_xor(vs, o, 32);
  float rstd = rsqrtf(vs * (1.0f / HH) + LN_EPS);
#pragma unroll
  for (int i = 0; i < 8; ++i) {
    int c = i * 32 + lane;
    out[(size_t)row * HH + c] = (v[i] - mu) * rstd * gamma[c] + beta[c];
  }
}

// ---------------- host-side orchestration ----------------
extern "C" void kernel_launch(void* const* d_in, const int* in_sizes, int n_in,
                              void* d_out, int out_size, void* d_ws, size_t ws_size,
                              hipStream_t stream) {
  const float* x     = (const float*)d_in[0];
  const int*   ei    = (const int*)d_in[1];
  // d_in[2] edge_weights: unused by reference
  const float* Wq    = (const float*)d_in[3];
  const float* Wk    = (const float*)d_in[4];
  const float* Wv    = (const float*)d_in[5];
  const float* Wo    = (const float*)d_in[6];
  const float* gamma = (const float*)d_in[7];
  const float* beta  = (const float*)d_in[8];
  float* outp = (float*)d_out;

  // workspace layout (all offsets 256B-aligned)
  char* base = (char*)d_ws;
  size_t off = 0;
  __bf16* xbf   = (__bf16*)(base + off); off += (size_t)ROWS * HH * 2;        // 2 MB
  __bf16* wcat  = (__bf16*)(base + off); off += (size_t)3 * HH * HH * 2;      // 384 KB
  __bf16* wobf  = (__bf16*)(base + off); off += (size_t)HH * HH * 2;          // 128 KB
  float*  qkv   = (float*) (base + off); off += (size_t)ROWS * 768 * 4;       // 12 MB
  unsigned int* maskbits = (unsigned int*)(base + off); off += (size_t)NN * (NN/32) * 4; // 512 KB
  int*    deg   = (int*)   (base + off); off += (size_t)NN * 4;               // 8 KB
  int*    adj   = (int*)   (base + off); off += (size_t)NN * CAP * 4;         // 2 MB
  __bf16* attnbf= (__bf16*)(base + off); off += (size_t)ROWS * HH * 2;        // 2 MB
  float*  ybuf  = (float*) (base + off); off += (size_t)ROWS * HH * 4;        // 4 MB

  // 1) convert inputs to bf16 (x, Wq|Wk|Wv concatenated, Wo)
  cvt_bf16_kernel<<<512, 256, 0, stream>>>(x, xbf, ROWS * HH);
  cvt_bf16_kernel<<<128, 256, 0, stream>>>(Wq, wcat,             HH * HH);
  cvt_bf16_kernel<<<128, 256, 0, stream>>>(Wk, wcat + HH * HH,   HH * HH);
  cvt_bf16_kernel<<<128, 256, 0, stream>>>(Wv, wcat + 2*HH*HH,   HH * HH);
  cvt_bf16_kernel<<<128, 256, 0, stream>>>(Wo, wobf,             HH * HH);

  // 2) adjacency: zero mask+deg (contiguous), then dedup-build capped lists
  zero_kernel<<<256, 256, 0, stream>>>(maskbits, NN * (NN / 32) + NN);
  edge_kernel<<<EDGES / 256, 256, 0, stream>>>(ei, maskbits, deg, adj);

  // 3) fused QKV projection: [4096,256] x [768,256]^T -> [4096,768] via WMMA
  {
    int waves = (ROWS / 16) * (768 / 16);        // 12288
    gemm_bf16_wmma_kernel<<<waves / 8, 256, 0, stream>>>(xbf, wcat, nullptr, qkv, 768, HH);
  }

  // 4) sparse masked attention -> bf16 [4096,256]
  attn_kernel<<<(BN * NHEAD * NN) / 8, 256, 0, stream>>>(qkv, deg, adj, attnbf);

  // 5) output projection + residual: [4096,256] x [256,256]^T + x -> ybuf
  {
    int waves = (ROWS / 16) * (HH / 16);         // 4096
    gemm_bf16_wmma_kernel<<<waves / 8, 256, 0, stream>>>(attnbf, wobf, x, ybuf, HH, HH);
  }

  // 6) LayerNorm -> d_out
  ln_kernel<<<ROWS / 8, 256, 0, stream>>>(ybuf, gamma, beta, outp);
  (void)in_sizes; (void)n_in; (void)out_size; (void)ws_size;
}
